// TransMIL_32212254720240
// MI455X (gfx1250) — compile-verified, matrix operand order checked by reading, not run on hardware
//
#include <hip/hip_runtime.h>
#include <hip/hip_bf16.h>
#include <cstdint>

// ---------------- CDNA5 WMMA types ----------------
typedef __attribute__((ext_vector_type(16))) _Float16 v16h;
typedef __attribute__((ext_vector_type(8)))  _Float16 v8h;
typedef __attribute__((ext_vector_type(8)))  float    v8f;

#define HEADS 8
#define DHEAD 64
#define LMARK 256
#define NPAD  16640   // 16385 padded front to multiple of 256
#define N1    16385   // 1 cls + 16384 instances
#define CDIM  512
#define DIN   1024
#define HGRID 128

union HV { v16h v; v8h h2[2]; _Float16 e[16]; };

// ---- optional CDNA5 async global->LDS path (guarded; falls back to ds_store) ----
#if defined(__has_builtin)
#if __has_builtin(__builtin_amdgcn_global_load_async_to_lds_b128) && \
    __has_builtin(__builtin_amdgcn_s_wait_asynccnt)
#define USE_ASYNC_LDS 1
#endif
#endif
#ifndef USE_ASYNC_LDS
#define USE_ASYNC_LDS 0
#endif

#if USE_ASYNC_LDS
typedef int v4i __attribute__((vector_size(16)));
typedef __attribute__((address_space(1))) v4i gv4i;   // global int4
typedef __attribute__((address_space(3))) v4i lv4i;   // LDS int4

__device__ __forceinline__ void async_copy_b128(const _Float16* gsrc, _Float16* ldst) {
  // generic LDS pointer carries the LDS byte offset in its low 32 bits (aperture rule)
  __builtin_amdgcn_global_load_async_to_lds_b128(
      (gv4i*)(uintptr_t)gsrc,
      (lv4i*)(uint32_t)(uintptr_t)ldst,
      0, 0);
}
#endif

// ---------------- elementwise helpers ----------------
__global__ void cast_f16_kernel(const float* __restrict__ in, _Float16* __restrict__ out, long n) {
  long i = (long)blockIdx.x * blockDim.x + threadIdx.x;
  if (i < n) out[i] = (_Float16)in[i];
}

__global__ void fill_zero_h_kernel(_Float16* __restrict__ p, long n) {
  long i = (long)blockIdx.x * blockDim.x + threadIdx.x;
  if (i < n) p[i] = (_Float16)0.0f;
}

__global__ void copy_f32_kernel(const float* __restrict__ in, float* __restrict__ out, long n) {
  long i = (long)blockIdx.x * blockDim.x + threadIdx.x;
  if (i < n) out[i] = in[i];
}

__global__ void add_inplace_kernel(float* __restrict__ a, const float* __restrict__ b, long n) {
  long i = (long)blockIdx.x * blockDim.x + threadIdx.x;
  if (i < n) a[i] += b[i];
}

// ---------------- WMMA strided-batched GEMM ----------------
// C[M,N] = alpha * A[M,K] * op(B) + bias ; A row-major (lda), per-batch stride sA.
// TB=1: B is N x K row-major (weight layout, computes A*B^T).
// TB=0: B is K x N row-major (computes A*B).
// Block = 256 threads = 8 waves; block tile 128M x 64N; each wave owns 16M x 64N
// (4 accumulators, A-fragment reused 4x with WMMA reuse-A hints).
// B tile (32K x 64N) staged through LDS each K-step:
//   TB=1: 4 waves store lane-fragment layout (b128 LDS loads on read),
//         via async global->LDS when available.
//   TB=0: all 256 threads do coalesced v8h global loads, store transposed
//         ([n][k], pitch 40 halves) so reads are two ds_load_b128 per fragment.
// C (f32) and Ch (f16 copy) each optional. Requires K%32==0, N%64==0, M%16==0.
template <int TB>
__global__ void gemm_wmma_kernel(
    const _Float16* __restrict__ A, long lda, long sA,
    const _Float16* __restrict__ B, long ldb, long sB,
    float* __restrict__ C, _Float16* __restrict__ Ch, long ldc, long sC,
    const float* __restrict__ bias, float alpha, int relu,
    int M, int N, int K)
{
  __shared__ __align__(16) _Float16 bsh[2560];
  const int  lane  = threadIdx.x & 31;
  const int  wave  = threadIdx.x >> 5;
  const long batch = blockIdx.z;
  A += batch * sA;
  B += batch * sB;
  const int  tileM  = blockIdx.y * 128 + wave * 16;
  const int  tileN0 = blockIdx.x * 64;
  const int  m = lane & 15;     // row (A) / col (B,C) within 16-wide tile
  const int  g = lane >> 4;     // K-half group
  const bool active = (tileM < M);
  const int  rowA = active ? (tileM + m) : m;

  v8f acc0 = {}, acc1 = {}, acc2 = {}, acc3 = {};

  auto ldT = [&](int j) {            // fragment from per-subtile layout
    HV b;
    b.h2[0] = *(const v8h*)&bsh[j * 512 + lane * 16];
    b.h2[1] = *(const v8h*)&bsh[j * 512 + lane * 16 + 8];
    return b;
  };
  auto ldN = [&](int j) {            // fragment from transposed [n][k] layout
    const int n = j * 16 + m;
    HV b;
    b.h2[0] = *(const v8h*)&bsh[n * 40 + g * 8];
    b.h2[1] = *(const v8h*)&bsh[n * 40 + 16 + g * 8];
    return b;
  };

  for (int kb = 0; kb < K; kb += 32) {
    __syncthreads();                 // previous iteration's LDS reads done
    if (TB) {
      if (wave < 4) {                // wave j stages subtile j's lane fragments
        const _Float16* bp = B + (long)(tileN0 + wave * 16 + m) * ldb + kb + g * 8;
        _Float16* dst = &bsh[wave * 512 + lane * 16];
#if USE_ASYNC_LDS
        async_copy_b128(bp, dst);
        async_copy_b128(bp + 16, dst + 8);
        __builtin_amdgcn_s_wait_asynccnt(0);
#else
        *(v8h*)dst       = *(const v8h*)bp;
        *(v8h*)(dst + 8) = *(const v8h*)(bp + 16);
#endif
      }
    } else {
      const int k  = threadIdx.x >> 3;   // 0..31
      const int pt = threadIdx.x & 7;    // 0..7 -> 8 n-values each
      const _Float16* bp = B + (long)(kb + k) * ldb + tileN0 + pt * 8;
      v8h bv = *(const v8h*)bp;          // coalesced 16B global load
#pragma unroll
      for (int q = 0; q < 8; ++q)
        bsh[(pt * 8 + q) * 40 + k] = bv[q];
    }
    __syncthreads();

    HV a;
    const _Float16* ap = A + (long)rowA * lda + kb + g * 8;
    a.h2[0] = *(const v8h*)ap;
    a.h2[1] = *(const v8h*)(ap + 16);
    __builtin_prefetch(ap + 64, 0, 1);   // global_prefetch_b8 on next A tile

    HV b0, b1, b2, b3;
    if (TB) { b0 = ldT(0); b1 = ldT(1); b2 = ldT(2); b3 = ldT(3); }
    else    { b0 = ldN(0); b1 = ldN(1); b2 = ldN(2); b3 = ldN(3); }

    acc0 = __builtin_amdgcn_wmma_f32_16x16x32_f16(false, a.v, false, b0.v,
                                                  (short)0, acc0, false, false);
    acc1 = __builtin_amdgcn_wmma_f32_16x16x32_f16(false, a.v, false, b1.v,
                                                  (short)0, acc1, true, false);
    acc2 = __builtin_amdgcn_wmma_f32_16x16x32_f16(false, a.v, false, b2.v,
                                                  (short)0, acc2, true, false);
    acc3 = __builtin_amdgcn_wmma_f32_16x16x32_f16(false, a.v, false, b3.v,
                                                  (short)0, acc3, true, false);
  }

  if (active) {
    v8f* accs[4] = { &acc0, &acc1, &acc2, &acc3 };
#pragma unroll
    for (int j = 0; j < 4; ++j) {
      const int col = tileN0 + j * 16 + m;
      v8f acc = *accs[j];
#pragma unroll
      for (int r = 0; r < 8; ++r) {
        const long row = tileM + r + 8 * g;
        float val = alpha * acc[r];
        if (bias) val += bias[col];
        if (relu) val = fmaxf(val, 0.0f);
        const long off = batch * sC + row * ldc + col;
        if (C)  C[off]  = val;
        if (Ch) Ch[off] = (_Float16)val;
      }
    }
  }
}

// ---------------- LayerNorm over rows of `cols` (writes f16) ----------------
__global__ void layernorm_kernel(const float* __restrict__ in, const float* __restrict__ g,
                                 const float* __restrict__ b, _Float16* __restrict__ outh, int cols)
{
  __shared__ float red[256];
  const long  row = blockIdx.x;
  const float* x  = in + row * (long)cols;
  const int   t   = threadIdx.x;
  float s = 0.0f;
  for (int c = t; c < cols; c += 256) s += x[c];
  red[t] = s; __syncthreads();
  for (int o = 128; o > 0; o >>= 1) { if (t < o) red[t] += red[t + o]; __syncthreads(); }
  const float mu = red[0] / cols;
  __syncthreads();
  float v = 0.0f;
  for (int c = t; c < cols; c += 256) { float d = x[c] - mu; v += d * d; }
  red[t] = v; __syncthreads();
  for (int o = 128; o > 0; o >>= 1) { if (t < o) red[t] += red[t + o]; __syncthreads(); }
  const float rstd = rsqrtf(red[0] / cols + 1e-5f);
  _Float16* o = outh + row * (long)cols;
  for (int c = t; c < cols; c += 256)
    o[c] = (_Float16)((x[c] - mu) * rstd * g[c] + b[c]);
}

// ---------------- row softmax (optional f32 + f16 outputs) ----------------
__global__ void softmax_kernel(const float* in, float* outf, _Float16* outh, int cols)
{
  __shared__ float red[256];
  const long  row = blockIdx.x;
  const float* x  = in + row * (long)cols;
  const int   t   = threadIdx.x;
  float mx = -3.4e38f;
  for (int c = t; c < cols; c += 256) mx = fmaxf(mx, x[c]);
  red[t] = mx; __syncthreads();
  for (int o = 128; o > 0; o >>= 1) { if (t < o) red[t] = fmaxf(red[t], red[t + o]); __syncthreads(); }
  mx = red[0]; __syncthreads();
  float s = 0.0f;
  for (int c = t; c < cols; c += 256) s += __expf(x[c] - mx);
  red[t] = s; __syncthreads();
  for (int o = 128; o > 0; o >>= 1) { if (t < o) red[t] += red[t + o]; __syncthreads(); }
  const float inv = 1.0f / red[0];
  for (int c = t; c < cols; c += 256) {
    float e = __expf(x[c] - mx) * inv;
    if (outf) outf[row * (long)cols + c] = e;
    if (outh) outh[row * (long)cols + c] = (_Float16)e;
  }
}

// ---------------- landmark means: out[h][j][d] = mean_t QKV[j*65+t][colOff+h*64+d] ----------------
__global__ void landmark_kernel(const float* __restrict__ QKV, int colOff, _Float16* __restrict__ out)
{
  int h = blockIdx.x >> 8;
  int j = blockIdx.x & 255;
  int d = threadIdx.x;            // blockDim = 64
  const float* p = QKV + (long)j * 65 * 1536 + colOff + h * 64 + d;
  float s = 0.0f;
  for (int tt = 0; tt < 65; ++tt) { s += *p; p += 1536; }
  out[((long)(h * 256 + j)) * 64 + d] = (_Float16)(s * (1.0f / 65.0f));
}

// ---------------- pinv init helpers ----------------
__global__ void rowsum_abs_kernel(const float* __restrict__ x, float* __restrict__ rs)
{
  __shared__ float red[256];
  const long row = blockIdx.x;              // 2048 rows of 256
  const int  t   = threadIdx.x;
  red[t] = fabsf(x[row * 256 + t]); __syncthreads();
  for (int o = 128; o > 0; o >>= 1) { if (t < o) red[t] += red[t + o]; __syncthreads(); }
  if (t == 0) rs[row] = red[0];
}

__global__ void colsum_abs_kernel(const float* __restrict__ x, float* __restrict__ cs)
{
  int h = blockIdx.x, c = threadIdx.x;      // grid 8, block 256
  float s = 0.0f;
  for (int i = 0; i < 256; ++i) s += fabsf(x[((long)h * 256 + i) * 256 + c]);
  cs[h * 256 + c] = s;
}

__global__ void maxmul_kernel(const float* __restrict__ rs, const float* __restrict__ cs,
                              float* __restrict__ out)
{
  __shared__ float r1[256], r2[256];
  int t = threadIdx.x;
  float a = -3.4e38f, b = -3.4e38f;
  for (int i = t; i < 2048; i += 256) { a = fmaxf(a, rs[i]); b = fmaxf(b, cs[i]); }
  r1[t] = a; r2[t] = b; __syncthreads();
  for (int o = 128; o > 0; o >>= 1) {
    if (t < o) { r1[t] = fmaxf(r1[t], r1[t + o]); r2[t] = fmaxf(r2[t], r2[t + o]); }
    __syncthreads();
  }
  if (t == 0) out[0] = r1[0] * r2[0];
}

__global__ void ztranspose_kernel(const float* __restrict__ x, const float* __restrict__ denom,
                                  _Float16* __restrict__ zh)
{
  long idx = (long)blockIdx.x * 256 + threadIdx.x;   // 8*256*256
  if (idx >= 8L * 256 * 256) return;
  long j = idx & 255; long hi = idx >> 8; long i = hi & 255; long h = hi >> 8;
  zh[idx] = (_Float16)(x[(h << 16) + (j << 8) + i] / denom[0]);
}

__global__ void diag_sub_kernel(float alpha, const float* __restrict__ in,
                                _Float16* __restrict__ out, long n)
{
  long idx = (long)blockIdx.x * 256 + threadIdx.x;
  if (idx >= n) return;
  int j = idx & 255, i = (idx >> 8) & 255;
  out[idx] = (_Float16)(((i == j) ? alpha : 0.0f) - in[idx]);
}

// ---------------- residual depthwise conv (33 taps along sequence) ----------------
__global__ void res_conv_kernel(const float* __restrict__ QKV, const float* __restrict__ rw,
                                float* __restrict__ O)
{
  long idx = (long)blockIdx.x * 256 + threadIdx.x;   // HEADS*NPAD*DHEAD
  if (idx >= (long)HEADS * NPAD * DHEAD) return;
  int  d  = idx & 63;
  long nd = idx >> 6;
  int  n  = (int)(nd % NPAD);
  int  h  = (int)(nd / NPAD);
  float s = 0.0f;
  for (int tp = 0; tp < 33; ++tp) {
    int src = n + tp - 16;
    if (src >= 0 && src < NPAD)
      s += rw[h * 33 + tp] * QKV[(long)src * 1536 + 1024 + h * 64 + d];
  }
  O[idx] += s;
}

__global__ void merge_heads_kernel(const float* __restrict__ O, _Float16* __restrict__ out)
{
  long idx = (long)blockIdx.x * 256 + threadIdx.x;   // NPAD*512
  if (idx >= (long)NPAD * 512) return;
  int  c = idx & 511;
  long n = idx >> 9;
  int  h = c >> 6, d = c & 63;
  out[idx] = (_Float16)O[((long)h * NPAD + n) * 64 + d];
}

// ---------------- PPEG: f + dw7 + dw5 + dw3 on 128x128x512 ----------------
__global__ void ppeg_kernel(const float* __restrict__ Hin,
                            const float* __restrict__ w7, const float* __restrict__ b7,
                            const float* __restrict__ w5, const float* __restrict__ b5,
                            const float* __restrict__ w3, const float* __restrict__ b3,
                            float* __restrict__ Hout)
{
  long idx = (long)blockIdx.x * 256 + threadIdx.x;   // 512*128*128
  if (idx >= (long)CDIM * HGRID * HGRID) return;
  int  c  = idx & 511;
  long sp = idx >> 9;
  int  xx = (int)(sp & 127);
  int  yy = (int)(sp >> 7);
  auto F = [&](int y, int x) -> float {
    if (y < 0 || y >= HGRID || x < 0 || x >= HGRID) return 0.0f;
    return Hin[(long)(1 + y * HGRID + x) * CDIM + c];
  };
  float acc = F(yy, xx) + b7[c] + b5[c] + b3[c];
  const float* W7 = w7 + (long)c * 49;
  const float* W5 = w5 + (long)c * 25;
  const float* W3 = w3 + (long)c * 9;
  for (int dy = -3; dy <= 3; ++dy)
    for (int dx = -3; dx <= 3; ++dx)
      acc += W7[(dy + 3) * 7 + dx + 3] * F(yy + dy, xx + dx);
  for (int dy = -2; dy <= 2; ++dy)
    for (int dx = -2; dx <= 2; ++dx)
      acc += W5[(dy + 2) * 5 + dx + 2] * F(yy + dy, xx + dx);
  for (int dy = -1; dy <= 1; ++dy)
    for (int dx = -1; dx <= 1; ++dx)
      acc += W3[(dy + 1) * 3 + dx + 1] * F(yy + dy, xx + dx);
  Hout[(long)(1 + yy * HGRID + xx) * CDIM + c] = acc;
}

// ---------------- final LN(row0) + 512->2 classifier ----------------
__global__ void final_head_kernel(const float* __restrict__ h0, const float* __restrict__ g,
                                  const float* __restrict__ bb, const float* __restrict__ w,
                                  const float* __restrict__ bias, float* __restrict__ out)
{
  __shared__ float red[512];
  int t = threadIdx.x;                 // blockDim = 512
  float x = h0[t];
  red[t] = x; __syncthreads();
  for (int o = 256; o > 0; o >>= 1) { if (t < o) red[t] += red[t + o]; __syncthreads(); }
  float mu = red[0] / 512.0f; __syncthreads();
  float d = x - mu;
  red[t] = d * d; __syncthreads();
  for (int o = 256; o > 0; o >>= 1) { if (t < o) red[t] += red[t + o]; __syncthreads(); }
  float rstd = rsqrtf(red[0] / 512.0f + 1e-5f); __syncthreads();
  float xn = d * rstd * g[t] + bb[t];
  for (int cls = 0; cls < 2; ++cls) {
    red[t] = xn * w[cls * 512 + t]; __syncthreads();
    for (int o = 256; o > 0; o >>= 1) { if (t < o) red[t] += red[t + o]; __syncthreads(); }
    if (t == 0) out[cls] = red[0] + bias[cls];
    __syncthreads();
  }
}

// ---------------- host-side orchestration ----------------
static inline char* bump(char*& p, size_t bytes) {
  char* r = p;
  p += (bytes + 255) & ~(size_t)255;
  return r;
}

static void launch_gemm(hipStream_t st, int TB,
                        const _Float16* A, long lda, long sA,
                        const _Float16* B, long ldb, long sB,
                        float* C, _Float16* Ch, long ldc, long sC,
                        const float* bias, float alpha, int relu,
                        int M, int N, int K, int batch)
{
  dim3 grid((unsigned)(N / 64), (unsigned)((M + 127) / 128), (unsigned)batch);
  if (TB)
    gemm_wmma_kernel<1><<<grid, 256, 0, st>>>(A, lda, sA, B, ldb, sB, C, Ch, ldc, sC,
                                              bias, alpha, relu, M, N, K);
  else
    gemm_wmma_kernel<0><<<grid, 256, 0, st>>>(A, lda, sA, B, ldb, sB, C, Ch, ldc, sC,
                                              bias, alpha, relu, M, N, K);
}

struct AttnWs {
  _Float16 *LNh, *QKVh, *qlh, *klh, *A1h, *A3h, *A2h, *A3Vh, *Zh0, *Zh1, *XZh, *TAh, *ZAh, *OCATh;
  float    *QKV, *S, *A2f, *rs, *cs, *denom, *XZf, *TBf, *Ohead, *ATTOUT;
};

static void attn_block(hipStream_t st, float* H,
                       const float* lng, const float* lnb,
                       const _Float16* wqkv, const _Float16* wout,
                       const float* outb, const float* resw, const AttnWs& w)
{
  const long nzz = 8L * 256 * 256;
  // zero front pad, layernorm into padded f16 buffer
  fill_zero_h_kernel<<<(255 * 512 + 255) / 256, 256, 0, st>>>(w.LNh, 255L * 512);
  layernorm_kernel<<<N1, 256, 0, st>>>(H, lng, lnb, w.LNh + 255L * 512, 512);
  // qkv = xp @ Wqkv^T
  launch_gemm(st, 1, w.LNh, 512, 0, wqkv, 512, 0, w.QKV, w.QKVh, 1536, 0,
              nullptr, 1.0f, 0, NPAD, 1536, 512, 1);
  // landmarks (unscaled q; 0.125 folded into score alphas)
  landmark_kernel<<<8 * 256, 64, 0, st>>>(w.QKV, 0,   w.qlh);
  landmark_kernel<<<8 * 256, 64, 0, st>>>(w.QKV, 512, w.klh);
  // a1 = softmax(0.125 * q @ kl^T)
  launch_gemm(st, 1, w.QKVh, 1536, 64, w.klh, 64, (long)LMARK * 64,
              w.S, nullptr, 256, (long)NPAD * 256, nullptr, 0.125f, 0, NPAD, 256, 64, 8);
  softmax_kernel<<<8 * NPAD, 256, 0, st>>>(w.S, nullptr, w.A1h, 256);
  // a2 = softmax(0.125 * ql @ kl^T)  (keep f32 for pinv scaling + f16 for GEMMs)
  launch_gemm(st, 1, w.qlh, 64, (long)LMARK * 64, w.klh, 64, (long)LMARK * 64,
              w.A2f, nullptr, 256, 65536, nullptr, 0.125f, 0, 256, 256, 64, 8);
  softmax_kernel<<<8 * 256, 256, 0, st>>>(w.A2f, w.A2f, w.A2h, 256);
  // a3 = softmax(0.125 * ql @ k^T)   (reuse S buffer)
  launch_gemm(st, 1, w.qlh, 64, (long)LMARK * 64, w.QKVh + 512, 1536, 64,
              w.S, nullptr, NPAD, (long)256 * NPAD, nullptr, 0.125f, 0, 256, NPAD, 64, 8);
  softmax_kernel<<<8 * 256, 256, 0, st>>>(w.S, nullptr, w.A3h, NPAD);
  // a3 @ v  -> (8,256,64) f16
  launch_gemm(st, 0, w.A3h, NPAD, (long)256 * NPAD, w.QKVh + 1024, 1536, 64,
              nullptr, w.A3Vh, 64, (long)LMARK * 64, nullptr, 1.0f, 0, 256, 64, NPAD, 8);
  // ---- pinv (Newton) on a2 ----
  rowsum_abs_kernel<<<2048, 256, 0, st>>>(w.A2f, w.rs);
  colsum_abs_kernel<<<8, 256, 0, st>>>(w.A2f, w.cs);
  maxmul_kernel<<<1, 256, 0, st>>>(w.rs, w.cs, w.denom);
  ztranspose_kernel<<<(unsigned)(nzz / 256), 256, 0, st>>>(w.A2f, w.denom, w.Zh0);
  _Float16* zc[2] = { w.Zh0, w.Zh1 };
  int cur = 0;
  for (int it = 0; it < 6; ++it) {
    launch_gemm(st, 0, w.A2h, 256, 65536, zc[cur], 256, 65536,
                w.XZf, w.XZh, 256, 65536, nullptr, 1.0f, 0, 256, 256, 256, 8);
    diag_sub_kernel<<<(unsigned)(nzz / 256), 256, 0, st>>>(7.0f, w.XZf, w.TAh, nzz);
    launch_gemm(st, 0, w.XZh, 256, 65536, w.TAh, 256, 65536,
                w.TBf, nullptr, 256, 65536, nullptr, 1.0f, 0, 256, 256, 256, 8);
    diag_sub_kernel<<<(unsigned)(nzz / 256), 256, 0, st>>>(15.0f, w.TBf, w.TAh, nzz);
    launch_gemm(st, 0, w.XZh, 256, 65536, w.TAh, 256, 65536,
                w.TBf, nullptr, 256, 65536, nullptr, 1.0f, 0, 256, 256, 256, 8);
    diag_sub_kernel<<<(unsigned)(nzz / 256), 256, 0, st>>>(13.0f, w.TBf, w.TAh, nzz);
    launch_gemm(st, 0, zc[cur], 256, 65536, w.TAh, 256, 65536,
                nullptr, zc[1 - cur], 256, 65536, nullptr, 0.25f, 0, 256, 256, 256, 8);
    cur ^= 1;
  }
  // ZA = pinv(a2) @ (a3 @ v)
  launch_gemm(st, 0, zc[cur], 256, 65536, w.A3Vh, 64, (long)LMARK * 64,
              nullptr, w.ZAh, 64, (long)LMARK * 64, nullptr, 1.0f, 0, 256, 64, 256, 8);
  // O = a1 @ ZA
  launch_gemm(st, 0, w.A1h, 256, (long)NPAD * 256, w.ZAh, 64, (long)LMARK * 64,
              w.Ohead, nullptr, 64, (long)NPAD * 64, nullptr, 1.0f, 0, NPAD, 64, 256, 8);
  // residual depthwise conv on v, merge heads, output projection
  res_conv_kernel<<<(unsigned)((8L * NPAD * 64) / 256), 256, 0, st>>>(w.QKV, resw, w.Ohead);
  merge_heads_kernel<<<(unsigned)(((long)NPAD * 512) / 256), 256, 0, st>>>(w.Ohead, w.OCATh);
  launch_gemm(st, 1, w.OCATh, 512, 0, wout, 512, 0, w.ATTOUT, nullptr, 512, 0,
              outb, 1.0f, 0, NPAD, 512, 512, 1);
  // h += attn_out[:, -16385:]
  add_inplace_kernel<<<(unsigned)(((long)N1 * 512) / 256), 256, 0, st>>>(
      H, w.ATTOUT + 255L * 512, (long)N1 * 512);
}

extern "C" void kernel_launch(void* const* d_in, const int* in_sizes, int n_in,
                              void* d_out, int out_size, void* d_ws, size_t ws_size,
                              hipStream_t stream)
{
  (void)in_sizes; (void)n_in; (void)out_size; (void)ws_size;
  const float* X      = (const float*)d_in[0];
  const float* fc1_w  = (const float*)d_in[1];
  const float* fc1_b  = (const float*)d_in[2];
  const float* cls_tk = (const float*)d_in[3];
  const float* ln1_g  = (const float*)d_in[4];
  const float* ln1_b  = (const float*)d_in[5];
  const float* qkv1_w = (const float*)d_in[6];
  const float* out1_w = (const float*)d_in[7];
  const float* out1_b = (const float*)d_in[8];
  const float* res1_w = (const float*)d_in[9];
  const float* w7 = (const float*)d_in[10];
  const float* b7 = (const float*)d_in[11];
  const float* w5 = (const float*)d_in[12];
  const float* b5 = (const float*)d_in[13];
  const float* w3 = (const float*)d_in[14];
  const float* b3 = (const float*)d_in[15];
  const float* ln2_g  = (const float*)d_in[16];
  const float* ln2_b  = (const float*)d_in[17];
  const float* qkv2_w = (const float*)d_in[18];
  const float* out2_w = (const float*)d_in[19];
  const float* out2_b = (const float*)d_in[20];
  const float* res2_w = (const float*)d_in[21];
  const float* lnf_g  = (const float*)d_in[22];
  const float* lnf_b  = (const float*)d_in[23];
  const float* fc2_w  = (const float*)d_in[24];
  const float* fc2_b  = (const float*)d_in[25];
  float* OUT = (float*)d_out;

  // ---- workspace carve ----
  char* wp = (char*)d_ws;
  _Float16* Xh      = (_Float16*)bump(wp, 16384L * 1024 * 2);
  _Float16* w_fc1h  = (_Float16*)bump(wp, 512L * 1024 * 2);
  _Float16* w_qkv1h = (_Float16*)bump(wp, 1536L * 512 * 2);
  _Float16* w_out1h = (_Float16*)bump(wp, 512L * 512 * 2);
  _Float16* w_qkv2h = (_Float16*)bump(wp, 1536L * 512 * 2);
  _Float16* w_out2h = (_Float16*)bump(wp, 512L * 512 * 2);
  float* Hbuf  = (float*)bump(wp, (long)N1 * 512 * 4);
  float* Hbuf2 = (float*)bump(wp, (long)N1 * 512 * 4);
  AttnWs aw;
  aw.LNh   = (_Float16*)bump(wp, (long)NPAD * 512 * 2);
  aw.QKV   = (float*)   bump(wp, (long)NPAD * 1536 * 4);
  aw.QKVh  = (_Float16*)bump(wp, (long)NPAD * 1536 * 2);
  aw.qlh   = (_Float16*)bump(wp, 8L * 256 * 64 * 2);
  aw.klh   = (_Float16*)bump(wp, 8L * 256 * 64 * 2);
  aw.S     = (float*)   bump(wp, 8L * NPAD * 256 * 4);
  aw.A1h   = (_Float16*)bump(wp, 8L * NPAD * 256 * 2);
  aw.A3h   = (_Float16*)bump(wp, 8L * NPAD * 256 * 2);
  aw.A2f   = (float*)   bump(wp, 8L * 65536 * 4);
  aw.A2h   = (_Float16*)bump(wp, 8L * 65536 * 2);
  aw.A3Vh  = (_Float16*)bump(wp, 8L * 256 * 64 * 2);
  aw.rs    = (float*)   bump(wp, 2048 * 4);
  aw.cs    = (float*)   bump(wp, 2048 * 4);
  aw.denom = (float*)   bump(wp, 256);
  aw.Zh0   = (_Float16*)bump(wp, 8L * 65536 * 2);
  aw.Zh1   = (_Float16*)bump(wp, 8L * 65536 * 2);
  aw.XZf   = (float*)   bump(wp, 8L * 65536 * 4);
  aw.XZh   = (_Float16*)bump(wp, 8L * 65536 * 2);
  aw.TBf   = (float*)   bump(wp, 8L * 65536 * 4);
  aw.TAh   = (_Float16*)bump(wp, 8L * 65536 * 2);
  aw.ZAh   = (_Float16*)bump(wp, 8L * 256 * 64 * 2);
  aw.Ohead = (float*)   bump(wp, 8L * NPAD * 64 * 4);
  aw.OCATh = (_Float16*)bump(wp, (long)NPAD * 512 * 2);
  aw.ATTOUT= (float*)   bump(wp, (long)NPAD * 512 * 4);

  // ---- f16 casts of X and weights ----
  auto cast = [&](const float* in, _Float16* out, long n) {
    cast_f16_kernel<<<(unsigned)((n + 255) / 256), 256, 0, stream>>>(in, out, n);
  };
  cast(X,      Xh,      16384L * 1024);
  cast(fc1_w,  w_fc1h,  512L * 1024);
  cast(qkv1_w, w_qkv1h, 1536L * 512);
  cast(out1_w, w_out1h, 512L * 512);
  cast(qkv2_w, w_qkv2h, 1536L * 512);
  cast(out2_w, w_out2h, 512L * 512);

  // ---- h = relu(X @ fc1_w^T + b); rows 1..16384 of Hbuf, row 0 = cls token ----
  launch_gemm(stream, 1, Xh, 1024, 0, w_fc1h, 1024, 0, Hbuf + 512, nullptr, 512, 0,
              fc1_b, 1.0f, 1, 16384, 512, 1024, 1);
  copy_f32_kernel<<<2, 256, 0, stream>>>(cls_tk, Hbuf, 512);

  // ---- block 1 ----
  attn_block(stream, Hbuf, ln1_g, ln1_b, w_qkv1h, w_out1h, out1_b, res1_w, aw);

  // ---- PPEG ----
  ppeg_kernel<<<(unsigned)(((long)CDIM * HGRID * HGRID) / 256), 256, 0, stream>>>(
      Hbuf, w7, b7, w5, b5, w3, b3, Hbuf2);
  copy_f32_kernel<<<2, 256, 0, stream>>>(Hbuf, Hbuf2, 512);   // cls row passthrough

  // ---- block 2 ----
  attn_block(stream, Hbuf2, ln2_g, ln2_b, w_qkv2h, w_out2h, out2_b, res2_w, aw);

  // ---- final LN(row0) + classifier ----
  final_head_kernel<<<1, 512, 0, stream>>>(Hbuf2, lnf_g, lnf_b, fc2_w, fc2_b, OUT);
}
